// LRPBlock_86629490360851
// MI455X (gfx1250) — matrix operand order repaired and checked by reading, not compiled, not run on hardware
//
#include <hip/hip_runtime.h>

// ---------------------------------------------------------------------------
// MI455X (gfx1250) implementation of the LRP sparse-conv block.
// Memory-bound workload (~13GB gather traffic vs 145 GFLOP) -> stage all
// gathered activations as f16 (halves gather bytes), run every GEMM through
// v_wmma_f32_16x16x32_f16 with f32 accumulation, pre-shuffle weights into
// WMMA fragment-major layout so B operands are contiguous 16B loads from L2.
// This round: wave-uniform full-tile fast paths (no per-element EXEC churn)
// and hoisted, clamped neighbor-row pointers out of the k-loop.
// ---------------------------------------------------------------------------

typedef __attribute__((ext_vector_type(16))) _Float16 v16h;
typedef __attribute__((ext_vector_type(8)))  _Float16 v8h;
typedef __attribute__((ext_vector_type(8)))  float    v8f;

#define NPTS  400000
#define KOFF  27

__device__ __forceinline__ v8f zero8() {
  v8f z = {0.f,0.f,0.f,0.f,0.f,0.f,0.f,0.f};
  return z;
}

__device__ __forceinline__ v16h cat8(v8h lo, v8h hi) {
  return __builtin_shufflevector(lo, hi, 0,1,2,3,4,5,6,7,8,9,10,11,12,13,14,15);
}

// A-fragment (16x32 f16, ISA 7.12.2): lane L holds row M=L&15.
// kb = (L>=16)?8:0 ; halves 0..7 -> K = kb+0..7 ; halves 8..15 -> K = kb+16..23.
// Both chunks are 8 contiguous halves (16B) in a row-major f16 row.
__device__ __forceinline__ v16h load_afrag(const _Float16* __restrict__ row,
                                           int kblk, int lane) {
  const int kb = (lane >> 4) << 3;           // 0 or 8
  const _Float16* p = row + 32 * kblk + kb;
  v8h lo = *(const v8h*)(p);
  v8h hi = *(const v8h*)(p + 16);
  return cat8(lo, hi);
}

// B-fragments are pre-shuffled: 16 contiguous halves per (frag, lane).
__device__ __forceinline__ v16h load_bfrag(const _Float16* __restrict__ WF,
                                           int fragId, int lane) {
  const _Float16* q = WF + ((size_t)fragId * 32 + lane) * 16;
  v8h lo = *(const v8h*)(q);
  v8h hi = *(const v8h*)(q + 8);
  return cat8(lo, hi);
}

__device__ __forceinline__ v8f wmma16(v16h a, v16h b, v8f c) {
  return __builtin_amdgcn_wmma_f32_16x16x32_f16(false, a, false, b,
                                                (short)0, c, false, false);
}

// ---------------------------------------------------------------------------
// Weight prep: f32 (K, CIN, COUT) row-major -> f16 fragment-major.
// B-layout (ISA 7.12.5 B 64x16 pattern, per 32-K block): lane L -> col n=16t+(L&15),
// half h -> K = 32*kblk + 8*(L>>4) + (h<8 ? h : h+8).
// ---------------------------------------------------------------------------
__global__ void prep_frag(const float* __restrict__ W, _Float16* __restrict__ dst,
                          int K, int CINp, int COUTp) {
  const int T  = COUTp >> 4;
  const int KB = CINp >> 5;
  const int total = K * T * KB * 32;
  int id = blockIdx.x * blockDim.x + threadIdx.x;
  if (id >= total) return;
  const int lane = id & 31;
  int rest = id >> 5;
  const int kblk = rest % KB; rest /= KB;
  const int t = rest % T;
  const int k = rest / T;
  const int n = 16 * t + (lane & 15);
  _Float16* o = dst + (size_t)id * 16;
#pragma unroll
  for (int h = 0; h < 16; ++h) {
    const int kk = 32 * kblk + ((lane >> 4) << 3) + (h < 8 ? h : h + 8);
    o[h] = (_Float16)W[((size_t)k * CINp + kk) * COUTp + n];
  }
}

__global__ void f32_to_f16(const float* __restrict__ x, _Float16* __restrict__ y,
                           long total) {
  long i = (long)blockIdx.x * blockDim.x + threadIdx.x;
  const long stride = (long)gridDim.x * blockDim.x;
  for (; i < total; i += stride) y[i] = (_Float16)x[i];
}

__global__ void zerof(float* __restrict__ p, int n) {
  int i = blockIdx.x * blockDim.x + threadIdx.x;
  if (i < n) p[i] = 0.f;
}

// ---------------------------------------------------------------------------
// Gather-GEMM sparse conv: out[N,64] = sum_k gather(in,nbr[:,k]) @ W[k].
// One wave owns a 32x64 f32 output tile (2 m-tiles x 4 n-tiles, 64 acc VGPRs).
// KB = CIN/32 WMMA K-blocks per k-offset.
// ---------------------------------------------------------------------------
template<int KB>
__global__ void __launch_bounds__(256)
gconv_wmma(const _Float16* __restrict__ fin, const int* __restrict__ nbr,
           const _Float16* __restrict__ WF, float* __restrict__ out, int n) {
  const int lane = threadIdx.x & 31;
  const int wave = threadIdx.x >> 5;
  const int m0 = blockIdx.x * 256 + wave * 32;
  if (m0 >= n) return;                       // wave-uniform exit

  v8f acc[2][4];
#pragma unroll
  for (int mt = 0; mt < 2; ++mt)
#pragma unroll
    for (int t = 0; t < 4; ++t) acc[mt][t] = zero8();

  // Hoisted, clamped neighbor-row pointers (loop-invariant per mt/lane).
  const int* nbp[2];
#pragma unroll
  for (int mt = 0; mt < 2; ++mt) {
    const int r = m0 + mt * 16 + (lane & 15);
    nbp[mt] = nbr + (size_t)((r < n) ? r : 0) * KOFF;
  }

  const int rowHalves = 32 * KB;
  for (int k = 0; k < KOFF; ++k) {
    v16h bf[4 * KB];
#pragma unroll
    for (int t = 0; t < 4; ++t)
#pragma unroll
      for (int kb = 0; kb < KB; ++kb)
        bf[t * KB + kb] = load_bfrag(WF, (k * 4 + t) * KB + kb, lane);

#pragma unroll
    for (int mt = 0; mt < 2; ++mt) {
      const unsigned idx = (unsigned)nbp[mt][k];
      const _Float16* row = fin + (size_t)idx * rowHalves;
#pragma unroll
      for (int kb = 0; kb < KB; ++kb) {
        v16h a = load_afrag(row, kb, lane);
#pragma unroll
        for (int t = 0; t < 4; ++t)
          acc[mt][t] = wmma16(a, bf[t * KB + kb], acc[mt][t]);
      }
    }
  }

  // C/D layout: VGPR j -> row m0+mt*16+8*(lane>=16)+j, col 16t+(lane&15).
  if (m0 + 32 <= n) {
    // Fast path: whole 32-row tile in range, branch-free stores.
#pragma unroll
    for (int mt = 0; mt < 2; ++mt) {
      const int rbase = m0 + mt * 16 + ((lane >> 4) << 3);
#pragma unroll
      for (int t = 0; t < 4; ++t) {
        const int nn = 16 * t + (lane & 15);
        float* o = out + (size_t)rbase * 64 + nn;
#pragma unroll
        for (int j = 0; j < 8; ++j) o[(size_t)j * 64] = acc[mt][t][j];
      }
    }
  } else {
#pragma unroll
    for (int mt = 0; mt < 2; ++mt) {
      const int rbase = m0 + mt * 16 + ((lane >> 4) << 3);
#pragma unroll
      for (int t = 0; t < 4; ++t) {
        const int nn = 16 * t + (lane & 15);
#pragma unroll
        for (int j = 0; j < 8; ++j) {
          const int r = rbase + j;
          if (r < n) out[(size_t)r * 64 + nn] = acc[mt][t][j];
        }
      }
    }
  }
}

// ---------------------------------------------------------------------------
// BatchNorm stats: per-channel sum / sumsq -> atomics into st[0..63], st[64..127]
// ---------------------------------------------------------------------------
__global__ void __launch_bounds__(256)
bnstats(const float* __restrict__ x, int n, float* __restrict__ st) {
  __shared__ float ls[256], lq[256];
  const int tid = threadIdx.x;
  const int c   = tid & 63;
  const int rg  = tid >> 6;                  // 4 row-groups per block
  const int rstride = gridDim.x << 2;
  float s = 0.f, q = 0.f;
  for (int r = (blockIdx.x << 2) + rg; r < n; r += rstride) {
    const float v = x[(size_t)r * 64 + c];
    s += v; q += v * v;
  }
  ls[tid] = s; lq[tid] = q;
  __syncthreads();
  if (tid < 128) { ls[tid] += ls[tid + 128]; lq[tid] += lq[tid + 128]; }
  __syncthreads();
  if (tid < 64) {
    s = ls[tid] + ls[tid + 64];
    q = lq[tid] + lq[tid + 64];
    atomicAdd(&st[c], s);
    atomicAdd(&st[64 + c], q);
  }
}

// ss[c] = rsqrt(var+eps)*g[c], ss[64+c] = b[c] - mu*ss[c]
__global__ void bnfinalize(const float* __restrict__ st, const float* __restrict__ g,
                           const float* __restrict__ b, float* __restrict__ ss,
                           float invn) {
  const int c = threadIdx.x;
  if (c < 64) {
    const float mu  = st[c] * invn;
    const float var = st[64 + c] * invn - mu * mu;
    const float sc  = rsqrtf(var + 1e-5f) * g[c];
    ss[c]      = sc;
    ss[64 + c] = b[c] - mu * sc;
  }
}

// relu(x*scale+shift) -> f16 activations for next gather stage
__global__ void bnrelu_h(const float* __restrict__ x, const float* __restrict__ ss,
                         _Float16* __restrict__ outh, long total) {
  long i = (long)blockIdx.x * blockDim.x + threadIdx.x;
  const long stride = (long)gridDim.x * blockDim.x;
  for (; i < total; i += stride) {
    const int c = (int)(i & 63);
    const float v = fmaxf(fmaf(x[i], ss[c], ss[64 + c]), 0.f);
    outh[i] = (_Float16)v;
  }
}

// ---------------------------------------------------------------------------
// BN2 + ReLU fused with residual 1x1 GEMM: x0 = relu(bn(out2)) + feats@Wlin.
// Writes x0 (f32, in-place over out2) and x0h (f16 for pooling / gates).
// ---------------------------------------------------------------------------
__global__ void __launch_bounds__(256)
bnrelu_resid(float* __restrict__ xio, const float* __restrict__ ss,
             const _Float16* __restrict__ fh, const _Float16* __restrict__ WlinF,
             _Float16* __restrict__ x0h, int n) {
  const int lane = threadIdx.x & 31;
  const int wave = threadIdx.x >> 5;
  const int m0 = blockIdx.x * 256 + wave * 32;
  if (m0 >= n) return;

  v8f acc[2][4];
#pragma unroll
  for (int mt = 0; mt < 2; ++mt)
#pragma unroll
    for (int t = 0; t < 4; ++t) acc[mt][t] = zero8();

  v16h bf[4];
#pragma unroll
  for (int t = 0; t < 4; ++t) bf[t] = load_bfrag(WlinF, t, lane);

#pragma unroll
  for (int mt = 0; mt < 2; ++mt) {
    const int r  = m0 + mt * 16 + (lane & 15);
    const int rc = (r < n) ? r : 0;
    v16h a = load_afrag(fh + (size_t)rc * 32, 0, lane);
#pragma unroll
    for (int t = 0; t < 4; ++t) acc[mt][t] = wmma16(a, bf[t], acc[mt][t]);
  }

  if (m0 + 32 <= n) {
#pragma unroll
    for (int mt = 0; mt < 2; ++mt) {
      const int rbase = m0 + mt * 16 + ((lane >> 4) << 3);
#pragma unroll
      for (int t = 0; t < 4; ++t) {
        const int nn = 16 * t + (lane & 15);
        const float sc = ss[nn], sh = ss[64 + nn];
        float*    io = xio + (size_t)rbase * 64 + nn;
        _Float16* oh = x0h + (size_t)rbase * 64 + nn;
#pragma unroll
        for (int j = 0; j < 8; ++j) {
          float v = fmaxf(fmaf(io[(size_t)j * 64], sc, sh), 0.f) + acc[mt][t][j];
          io[(size_t)j * 64] = v;
          oh[(size_t)j * 64] = (_Float16)v;
        }
      }
    }
  } else {
#pragma unroll
    for (int mt = 0; mt < 2; ++mt) {
      const int rbase = m0 + mt * 16 + ((lane >> 4) << 3);
#pragma unroll
      for (int t = 0; t < 4; ++t) {
        const int nn = 16 * t + (lane & 15);
        const float sc = ss[nn], sh = ss[64 + nn];
#pragma unroll
        for (int j = 0; j < 8; ++j) {
          const int r = rbase + j;
          if (r < n) {
            const size_t o = (size_t)r * 64 + nn;
            float v = fmaxf(fmaf(xio[o], sc, sh), 0.f) + acc[mt][t][j];
            xio[o] = v;
            x0h[o] = (_Float16)v;
          }
        }
      }
    }
  }
}

// ---------------------------------------------------------------------------
// Neighborhood average pool over 27 offsets (f16 in, f16 out). Pure gather:
// lane handles 2 packed channels (4B load), 32 lanes cover all 64 channels.
// ---------------------------------------------------------------------------
__global__ void pool27(const _Float16* __restrict__ in, const int* __restrict__ nbr,
                       _Float16* __restrict__ out, int n) {
  const int row = blockIdx.x * blockDim.y + threadIdx.y;
  if (row >= n) return;
  const int lane = threadIdx.x;
  const int* nb = nbr + (size_t)row * KOFF;
  float a0 = 0.f, a1 = 0.f;
  for (int k = 0; k < KOFF; ++k) {
    const unsigned idx = (unsigned)nb[k];
    union { unsigned u; _Float16 h[2]; } v;
    v.u = *(const unsigned*)(in + (size_t)idx * 64 + (lane << 1));
    a0 += (float)v.h[0];
    a1 += (float)v.h[1];
  }
  const float inv = 1.f / 27.f;
  union { unsigned u; _Float16 h[2]; } o;
  o.h[0] = (_Float16)(a0 * inv);
  o.h[1] = (_Float16)(a1 * inv);
  *(unsigned*)(out + (size_t)row * 64 + (lane << 1)) = o.u;
}

// ---------------------------------------------------------------------------
// Gates GEMM (x0h[16x64] @ Wsw[64x192], 24 WMMAs) fused with the 3-scale
// weighted combine:  y = s0*x0 + s1*(x1*g0 + x2*g1 + x3*g2).
// ---------------------------------------------------------------------------
__global__ void __launch_bounds__(256)
switch_combine(const _Float16* __restrict__ x0h, const float* __restrict__ x0f,
               const _Float16* __restrict__ x1h, const _Float16* __restrict__ x2h,
               const _Float16* __restrict__ x3h, const _Float16* __restrict__ WswF,
               const float* __restrict__ s0, const float* __restrict__ s1,
               float* __restrict__ out, int n) {
  const int lane = threadIdx.x & 31;
  const int wave = threadIdx.x >> 5;
  const int m0 = blockIdx.x * 128 + wave * 16;
  if (m0 >= n) return;

  const int r  = m0 + (lane & 15);
  const int rc = (r < n) ? r : 0;
  const _Float16* row = x0h + (size_t)rc * 64;
  v16h a0 = load_afrag(row, 0, lane);
  v16h a1 = load_afrag(row, 1, lane);

  v8f acc[12];
#pragma unroll
  for (int t = 0; t < 12; ++t) acc[t] = zero8();
#pragma unroll
  for (int t = 0; t < 12; ++t) {
    acc[t] = wmma16(a0, load_bfrag(WswF, t * 2 + 0, lane), acc[t]);
    acc[t] = wmma16(a1, load_bfrag(WswF, t * 2 + 1, lane), acc[t]);
  }

  // gates[:,0,c]=cols 0..63 (tiles 0..3), [:,1]=tiles 4..7, [:,2]=tiles 8..11
  const int rbase = m0 + ((lane >> 4) << 3);
  if (m0 + 16 <= n) {
#pragma unroll
    for (int t = 0; t < 4; ++t) {
      const int c = 16 * t + (lane & 15);
      const float sc0 = s0[c], sc1 = s1[c];
      const size_t ob = (size_t)rbase * 64 + c;
#pragma unroll
      for (int j = 0; j < 8; ++j) {
        const size_t o = ob + (size_t)j * 64;
        const float f2 = (float)x1h[o] * acc[t][j]
                       + (float)x2h[o] * acc[t + 4][j]
                       + (float)x3h[o] * acc[t + 8][j];
        out[o] = fmaf(sc0, x0f[o], sc1 * f2);
      }
    }
  } else {
#pragma unroll
    for (int t = 0; t < 4; ++t) {
      const int c = 16 * t + (lane & 15);
      const float sc0 = s0[c], sc1 = s1[c];
#pragma unroll
      for (int j = 0; j < 8; ++j) {
        const int rr = rbase + j;
        if (rr < n) {
          const size_t o = (size_t)rr * 64 + c;
          const float f2 = (float)x1h[o] * acc[t][j]
                         + (float)x2h[o] * acc[t + 4][j]
                         + (float)x3h[o] * acc[t + 8][j];
          out[o] = fmaf(sc0, x0f[o], sc1 * f2);
        }
      }
    }
  }
}

// ---------------------------------------------------------------------------
extern "C" void kernel_launch(void* const* d_in, const int* in_sizes, int n_in,
                              void* d_out, int out_size, void* d_ws, size_t ws_size,
                              hipStream_t stream) {
  const float* feats = (const float*)d_in[0];
  const int*   nbr1  = (const int*)d_in[1];
  const int*   nbr3  = (const int*)d_in[2];
  const int*   nbr9  = (const int*)d_in[3];
  const float* W0a   = (const float*)d_in[4];
  const float* g1    = (const float*)d_in[5];
  const float* b1    = (const float*)d_in[6];
  const float* W0b   = (const float*)d_in[7];
  const float* g2    = (const float*)d_in[8];
  const float* b2    = (const float*)d_in[9];
  const float* Wlin  = (const float*)d_in[10];
  const float* Wsw   = (const float*)d_in[11];
  const float* sc0   = (const float*)d_in[12];
  const float* sc1   = (const float*)d_in[13];
  float* out = (float*)d_out;
  const int N = NPTS;

  // -------- workspace carve-out (256B aligned) --------
  char* ws = (char*)d_ws;
  size_t off = 0;
  auto carve = [&](size_t bytes) -> void* {
    void* p = ws + off;
    off = (off + bytes + 255) & ~(size_t)255;
    return p;
  };
  _Float16* h_feats = (_Float16*)carve((size_t)N * 32 * 2);  // feats in f16
  _Float16* xh      = (_Float16*)carve((size_t)N * 64 * 2);  // xa_h, then x0h
  float*    xf      = (float*)   carve((size_t)N * 64 * 4);  // out1 -> out2 -> x0
  _Float16* x1h     = (_Float16*)carve((size_t)N * 64 * 2);
  _Float16* x2h     = (_Float16*)carve((size_t)N * 64 * 2);
  _Float16* x3h     = (_Float16*)carve((size_t)N * 64 * 2);
  _Float16* WaF     = (_Float16*)carve((size_t)KOFF * 4 * 1 * 32 * 16 * 2);
  _Float16* WbF     = (_Float16*)carve((size_t)KOFF * 4 * 2 * 32 * 16 * 2);
  _Float16* WlinF   = (_Float16*)carve((size_t)4 * 32 * 16 * 2);
  _Float16* WswF    = (_Float16*)carve((size_t)12 * 2 * 32 * 16 * 2);
  float*    st1     = (float*)carve(128 * 4);
  float*    st2     = (float*)carve(128 * 4);
  float*    ss1     = (float*)carve(128 * 4);
  float*    ss2     = (float*)carve(128 * 4);
  (void)ws_size; (void)in_sizes; (void)n_in; (void)out_size;

  const int convGrid = (N + 255) / 256;        // 256 rows / block (8 waves x 32)
  const float invN = 1.0f / (float)N;

  // -------- prep: zero stats, shuffle weights, convert feats --------
  zerof<<<1, 256, 0, stream>>>(st1, 128);
  zerof<<<1, 256, 0, stream>>>(st2, 128);
  prep_frag<<<(KOFF * 4 * 1 * 32 + 255) / 256, 256, 0, stream>>>(W0a, WaF, KOFF, 32, 64);
  prep_frag<<<(KOFF * 4 * 2 * 32 + 255) / 256, 256, 0, stream>>>(W0b, WbF, KOFF, 64, 64);
  prep_frag<<<1, 128, 0, stream>>>(Wlin, WlinF, 1, 32, 64);
  prep_frag<<<3, 256, 0, stream>>>(Wsw, WswF, 1, 64, 192);
  f32_to_f16<<<8192, 256, 0, stream>>>(feats, h_feats, (long)N * 32);

  // -------- conv block 1: conv -> BN stats -> BN+ReLU (f16 out) --------
  gconv_wmma<1><<<convGrid, 256, 0, stream>>>(h_feats, nbr1, WaF, xf, N);
  bnstats<<<512, 256, 0, stream>>>(xf, N, st1);
  bnfinalize<<<1, 64, 0, stream>>>(st1, g1, b1, ss1, invN);
  bnrelu_h<<<8192, 256, 0, stream>>>(xf, ss1, xh, (long)N * 64);

  // -------- conv block 2 (CIN=64) --------
  gconv_wmma<2><<<convGrid, 256, 0, stream>>>(xh, nbr1, WbF, xf, N);
  bnstats<<<512, 256, 0, stream>>>(xf, N, st2);
  bnfinalize<<<1, 64, 0, stream>>>(st2, g2, b2, ss2, invN);

  // -------- BN2+ReLU + residual GEMM -> x0 (f32 in-place) and x0h --------
  bnrelu_resid<<<convGrid, 256, 0, stream>>>(xf, ss2, h_feats, WlinF, xh, N);

  // -------- multi-scale pooling chain --------
  dim3 pblk(32, 8);
  const int pgrid = (N + 7) / 8;
  pool27<<<pgrid, pblk, 0, stream>>>(xh,  nbr1, x1h, N);
  pool27<<<pgrid, pblk, 0, stream>>>(x1h, nbr3, x2h, N);
  pool27<<<pgrid, pblk, 0, stream>>>(x2h, nbr9, x3h, N);

  // -------- gates GEMM + weighted combine --------
  switch_combine<<<(N + 127) / 128, 256, 0, stream>>>(xh, xf, x1h, x2h, x3h,
                                                      WswF, sc0, sc1, out, N);
}
